// LSTMClassifier_32624571580676
// MI455X (gfx1250) — compile-verified, matrix operand order checked by reading, not run on hardware
//
#include <hip/hip_runtime.h>
#include <hip/hip_bf16.h>

// ---------------------------------------------------------------------------
// 2-layer LSTM + FC head for MI455X (gfx1250, wave32, WMMA).
//
// The problem is a sequential chain of [64,K]x[K,1024] GEMMs; all tensors fit
// in L2 (192MB) so it is latency-bound, not bandwidth/FLOP bound. Design:
//  * gate/hidden split over 16 WGs per layer; each WG's Kx64 bf16 weight
//    slice lives in VGPRs (loaded once, reused 2048x); __launch_bounds__
//    (512,1) so nothing spills (verified round 2: no scratch in hot loop).
//  * BOTH layers run in ONE persistent kernel (32 WGs), software-pipelined:
//    at step s, layer1 computes t=s, layer2 computes t=s-1 -> 2049 grid
//    barriers instead of 4096.
//  * A = [x_t | h_{t-1}] staged in LDS (bf16, vectorized fill). A-fragments
//    are group-pipelined (2 ping-pong groups of 4 chunks, constant indices ->
//    distinct registers) so ds_load latency hides under v_wmma.
//  * monotonic atomic grid barrier + device fences per step.
// ---------------------------------------------------------------------------

typedef __attribute__((ext_vector_type(16))) __bf16 v16bf;
typedef __attribute__((ext_vector_type(8)))  float  v8f;

#define B_SZ      64
#define T_SZ      2048
#define FIN_SZ    104
#define H_SZ      256
#define NROWS     1024      // 4 gates * H
#define NWG_L     16        // workgroups per layer
#define NWG_TOT   32        // both layers, pipelined
#define JS        16        // hidden units per WG
#define NTHREADS  512       // 16 waves
#define KPAD1     384       // 104 + 256 padded to 12 chunks of 32
#define KPAD2     512       // 256 + 256 = 16 chunks of 32

__device__ __forceinline__ unsigned short f2bf(float f) {
  unsigned int u = __float_as_uint(f);
  u += 0x7FFFu + ((u >> 16) & 1u);          // round-to-nearest-even
  return (unsigned short)(u >> 16);
}
__device__ __forceinline__ unsigned int pk2bf(float lo, float hi) {
  return (unsigned int)f2bf(lo) | ((unsigned int)f2bf(hi) << 16);
}
__device__ __forceinline__ float bf2f(unsigned short s) {
  return __uint_as_float(((unsigned int)s) << 16);
}
__device__ __forceinline__ float sigmoid_(float x) {
  return 1.0f / (1.0f + __expf(-x));
}
__device__ __forceinline__ float tanh_(float x) {
  float e = __expf(-2.0f * x);
  return (1.0f - e) / (1.0f + e);
}

union Frag {       // 32 bytes per lane = one 16x16x32 bf16 WMMA operand frag
  uint4 q[2];
  v16bf v;
};

// --------------------------------------------------------------------------
// Weight prepack: Wcat[r][k] = [W_ih | W_hh] (bf16, zero-padded K), fused bias.
// --------------------------------------------------------------------------
__global__ void prepack_kernel(const float* __restrict__ Wih1, const float* __restrict__ Whh1,
                               const float* __restrict__ bih1, const float* __restrict__ bhh1,
                               const float* __restrict__ Wih2, const float* __restrict__ Whh2,
                               const float* __restrict__ bih2, const float* __restrict__ bhh2,
                               unsigned short* __restrict__ W1, unsigned short* __restrict__ W2,
                               float* __restrict__ Bs1, float* __restrict__ Bs2) {
  const int tid = blockIdx.x * blockDim.x + threadIdx.x;
  const int nth = gridDim.x * blockDim.x;
  for (int i = tid; i < NROWS * KPAD1; i += nth) {
    int r = i / KPAD1, k = i - r * KPAD1;
    float v = 0.0f;
    if (k < FIN_SZ)                 v = Wih1[r * FIN_SZ + k];
    else if (k < FIN_SZ + H_SZ)     v = Whh1[r * H_SZ + (k - FIN_SZ)];
    W1[i] = f2bf(v);
  }
  for (int i = tid; i < NROWS * KPAD2; i += nth) {
    int r = i / KPAD2, k = i - r * KPAD2;
    float v = (k < H_SZ) ? Wih2[r * H_SZ + k] : Whh2[r * H_SZ + (k - H_SZ)];
    W2[i] = f2bf(v);
  }
  for (int r = tid; r < NROWS; r += nth) {
    Bs1[r] = bih1[r] + bhh1[r];
    Bs2[r] = bih2[r] + bhh2[r];
  }
}

__global__ void init_kernel(unsigned int* bar) {
  if (threadIdx.x < 2) bar[threadIdx.x] = 0u;
}

// --------------------------------------------------------------------------
// Persistent per-layer body (inlined into the fused kernel). LAG = pipeline
// delay in global steps (layer1: 0, layer2: 1).
// --------------------------------------------------------------------------
template <int KPAD, bool LAYER1, int LAG>
__device__ __forceinline__ void lstm_persist(const float* __restrict__ x,
                                             const unsigned short* __restrict__ hin,
                                             const unsigned short* __restrict__ W,
                                             const float* __restrict__ bias,
                                             unsigned short* __restrict__ hseq,
                                             unsigned short* __restrict__ hbuf,
                                             unsigned int* __restrict__ bar,
                                             unsigned char* smem, int wg) {
  constexpr int KCH = KPAD / 32;
  constexpr int GRP = 4;            // A-fragment chunks per pipeline group
  constexpr int NG  = KCH / GRP;    // 3 (layer1) or 4 (layer2) groups
  unsigned short* Alds  = (unsigned short*)smem;                    // [64][KPAD] bf16
  float*          gates = (float*)(smem + B_SZ * KPAD * 2);         // [4][64][16]
  float*          cst   = gates + 4 * B_SZ * JS;                    // [64][16] cell state
  float*          bls   = cst + B_SZ * JS;                          // [4][16] bias slice

  const int tid   = threadIdx.x;
  const int lane  = tid & 31;
  const int wave  = tid >> 5;
  const int mtile = wave >> 2;      // 4 batch tiles of 16
  const int gtile = wave & 3;       // 4 gates
  const int ncol  = lane & 15;
  const int khalf = lane >> 4;

  // --- Preload this wave's B-matrix fragments into registers (once). ---
  Frag bfr[KCH];
  {
    const int r = gtile * H_SZ + wg * JS + ncol;
    const uint4* wrow = (const uint4*)(W + (size_t)r * KPAD);
#pragma unroll
    for (int kc = 0; kc < KCH; ++kc) {
      bfr[kc].q[0] = wrow[kc * 4 + khalf];       // K = kbase + 8*khalf .. +7
      bfr[kc].q[1] = wrow[kc * 4 + khalf + 2];   // K = kbase + 16 + 8*khalf .. +7
    }
  }

  // --- Init LDS: cell state, bias slice, zero pad region of A (layer1). ---
  for (int i = tid; i < B_SZ * JS; i += NTHREADS) cst[i] = 0.0f;
  if (tid < 4 * JS) bls[tid] = bias[(tid >> 4) * H_SZ + wg * JS + (tid & 15)];
  if (LAYER1) {  // k in [360,384) stays zero for the whole kernel
    for (int i = tid; i < B_SZ * (KPAD - FIN_SZ - H_SZ); i += NTHREADS) {
      int b = i / (KPAD - FIN_SZ - H_SZ), k = i - b * (KPAD - FIN_SZ - H_SZ);
      Alds[b * KPAD + FIN_SZ + H_SZ + k] = 0;
    }
  }
  // zero own slice of both h ping-pong buffers
  for (int i = tid; i < 2 * B_SZ; i += NTHREADS) {
    int buf = i >> 6, b = i & 63;
    for (int j = 0; j < JS; ++j)
      hbuf[buf * B_SZ * H_SZ + b * H_SZ + wg * JS + j] = 0;
  }
  __syncthreads();

  // --- Pre-loop grid barrier. ---
  if (tid == 0) {
    __threadfence();
    atomicAdd(bar, 1u);
    while (__hip_atomic_load(bar, __ATOMIC_RELAXED, __HIP_MEMORY_SCOPE_AGENT) <
           (unsigned int)NWG_TOT)
      __builtin_amdgcn_s_sleep(1);
  }
  __syncthreads();
  __threadfence();

  // Global pipeline steps: s = 0 .. T (inclusive). This layer computes
  // t = s - LAG when 0 <= t < T; otherwise it only participates in barriers.
  for (int s = 0; s <= T_SZ; ++s) {
    const int t = s - LAG;
    const bool active = (t >= 0) && (t < T_SZ);
    if (active) {
      const unsigned short* hprev = hbuf + (t & 1) * B_SZ * H_SZ;
      unsigned short*       hnext = hbuf + ((t + 1) & 1) * B_SZ * H_SZ;

      // ---- Phase A: build A = [x_t | h_{t-1}] (bf16) in LDS, vectorized ----
      if (LAYER1) {
        // x region: 64 rows x 26 float4 -> pack to 4 bf16 (uint2 stores)
        for (int i = tid; i < B_SZ * (FIN_SZ / 4); i += NTHREADS) {
          int b = i / (FIN_SZ / 4), g = i - b * (FIN_SZ / 4);
          const float4 f = *(const float4*)(x + ((size_t)b * T_SZ + t) * FIN_SZ + g * 4);
          uint2 p; p.x = pk2bf(f.x, f.y); p.y = pk2bf(f.z, f.w);
          *(uint2*)(Alds + b * KPAD + g * 4) = p;
          if (g == 0 && t + 1 < T_SZ)
            __builtin_prefetch(&x[((size_t)b * T_SZ + t + 1) * FIN_SZ], 0, 1);
        }
        // h region: 64 rows x 32 uint4 (offset 104 elems = 208B, 16B aligned)
        for (int i = tid; i < B_SZ * (H_SZ / 8); i += NTHREADS) {
          int b = i >> 5, g = i & 31;
          ((uint4*)(Alds + b * KPAD + FIN_SZ))[g] =
              ((const uint4*)(hprev + b * H_SZ))[g];
        }
      } else {
        // x region = h1[b][t][:], h region = hprev; both 32 uint4 per row
        for (int i = tid; i < 2 * B_SZ * (H_SZ / 8); i += NTHREADS) {
          int half = i >> 11;            // 0: h1_t, 1: hprev
          int rem  = i & 2047;
          int b = rem >> 5, g = rem & 31;
          const uint4 v = half == 0
              ? ((const uint4*)(hin + ((size_t)b * T_SZ + t) * H_SZ))[g]
              : ((const uint4*)(hprev + b * H_SZ))[g];
          ((uint4*)(Alds + b * KPAD + half * H_SZ))[g] = v;
        }
      }
      __syncthreads();

      // ---- Phase B: gates tile = A x Wslice^T via WMMA, group-pipelined ----
      // Two ping-pong groups of 4 A-fragments with compile-time indices ->
      // distinct registers; next group's ds_loads issue before the partial
      // wait on the current group, hiding LDS latency under the matrix pipe.
      v8f acc = {0.f, 0.f, 0.f, 0.f, 0.f, 0.f, 0.f, 0.f};
      const uint4* arow = (const uint4*)(Alds + (size_t)(mtile * 16 + ncol) * KPAD);
      Frag ag[2][GRP];
#pragma unroll
      for (int u = 0; u < GRP; ++u) {
        ag[0][u].q[0] = arow[u * 4 + khalf];
        ag[0][u].q[1] = arow[u * 4 + khalf + 2];
      }
#pragma unroll
      for (int g = 0; g < NG; ++g) {
        const int cur = g & 1, nxt = cur ^ 1;
        if (g + 1 < NG) {
#pragma unroll
          for (int u = 0; u < GRP; ++u) {
            ag[nxt][u].q[0] = arow[((g + 1) * GRP + u) * 4 + khalf];
            ag[nxt][u].q[1] = arow[((g + 1) * GRP + u) * 4 + khalf + 2];
          }
        }
#pragma unroll
        for (int u = 0; u < GRP; ++u) {
          acc = __builtin_amdgcn_wmma_f32_16x16x32_bf16(
              false, ag[cur][u].v, false, bfr[g * GRP + u].v,
              (short)0, acc, false, false);
        }
      }
      {
        // C layout: VGPR r, lane<16 -> M = r ; lane>=16 -> M = 8 + r.
        const int rbase = mtile * 16 + khalf * 8;
        float* gp = gates + (gtile * B_SZ + rbase) * JS + ncol;
#pragma unroll
        for (int r2 = 0; r2 < 8; ++r2) gp[r2 * JS] = acc[r2];
      }
      __syncthreads();

      // ---- Phase C: i,f,g,o nonlinearity + cell/hidden update ----
      for (int p = tid; p < B_SZ * JS; p += NTHREADS) {
        int b = p >> 4, j = p & 15;
        float gi = sigmoid_(gates[(0 * B_SZ + b) * JS + j] + bls[0 * JS + j]);
        float gf = sigmoid_(gates[(1 * B_SZ + b) * JS + j] + bls[1 * JS + j]);
        float gg = tanh_   (gates[(2 * B_SZ + b) * JS + j] + bls[2 * JS + j]);
        float go = sigmoid_(gates[(3 * B_SZ + b) * JS + j] + bls[3 * JS + j]);
        float c  = gf * cst[p] + gi * gg;
        cst[p] = c;
        unsigned short hb = f2bf(go * tanh_(c));
        hnext[b * H_SZ + wg * JS + j] = hb;
        hseq[((size_t)b * T_SZ + t) * H_SZ + wg * JS + j] = hb;
      }
      __syncthreads();
    }

    // ---- Grid barrier (monotonic counter; sense-free) ----
    if (tid == 0) {
      __threadfence();
      atomicAdd(bar, 1u);
      const unsigned int target = (unsigned int)NWG_TOT * (unsigned int)(s + 2);
      while (__hip_atomic_load(bar, __ATOMIC_RELAXED, __HIP_MEMORY_SCOPE_AGENT) < target)
        __builtin_amdgcn_s_sleep(1);
    }
    __syncthreads();
    __threadfence();
  }
}

// --------------------------------------------------------------------------
// Fused pipelined 2-layer LSTM. Grid = 32 WGs x 512 threads.
// WGs 0..15: layer 1 (t = s). WGs 16..31: layer 2 (t = s-1).
// --------------------------------------------------------------------------
__global__ __launch_bounds__(NTHREADS, 1)
void lstm2_kernel(const float* __restrict__ x,
                  const unsigned short* __restrict__ W1, const float* __restrict__ Bs1,
                  const unsigned short* __restrict__ W2, const float* __restrict__ Bs2,
                  unsigned short* __restrict__ h1, unsigned short* __restrict__ h2,
                  unsigned short* __restrict__ hbuf1, unsigned short* __restrict__ hbuf2,
                  unsigned int* __restrict__ bar) {
  extern __shared__ unsigned char smem[];
  const int wg = blockIdx.x;
  if (wg < NWG_L) {
    lstm_persist<KPAD1, true, 0>(x, nullptr, W1, Bs1, h1, hbuf1, bar, smem, wg);
  } else {
    lstm_persist<KPAD2, false, 1>(nullptr, h1, W2, Bs2, h2, hbuf2, bar, smem, wg - NWG_L);
  }
}

// --------------------------------------------------------------------------
// FC head: out[b,t,c] = h2[b,t,:] . W_fc[c,:] + b_fc[c]   (C = 2)
// --------------------------------------------------------------------------
__global__ void fc_kernel(const unsigned short* __restrict__ h2,
                          const float* __restrict__ Wfc,
                          const float* __restrict__ bfc,
                          float* __restrict__ out) {
  int idx = blockIdx.x * blockDim.x + threadIdx.x;
  if (idx >= B_SZ * T_SZ) return;
  const unsigned short* row = h2 + (size_t)idx * H_SZ;
  float a0 = bfc[0], a1 = bfc[1];
#pragma unroll 8
  for (int j = 0; j < H_SZ; ++j) {
    float v = bf2f(row[j]);
    a0 += v * Wfc[j];
    a1 += v * Wfc[H_SZ + j];
  }
  out[idx * 2 + 0] = a0;
  out[idx * 2 + 1] = a1;
}

// --------------------------------------------------------------------------
extern "C" void kernel_launch(void* const* d_in, const int* in_sizes, int n_in,
                              void* d_out, int out_size, void* d_ws, size_t ws_size,
                              hipStream_t stream) {
  const float* x    = (const float*)d_in[0];
  const float* Wih1 = (const float*)d_in[1];
  const float* Whh1 = (const float*)d_in[2];
  const float* bih1 = (const float*)d_in[3];
  const float* bhh1 = (const float*)d_in[4];
  const float* Wih2 = (const float*)d_in[5];
  const float* Whh2 = (const float*)d_in[6];
  const float* bih2 = (const float*)d_in[7];
  const float* bhh2 = (const float*)d_in[8];
  const float* Wfc  = (const float*)d_in[9];
  const float* bfc  = (const float*)d_in[10];
  float* out        = (float*)d_out;

  // Workspace carve (~136 MB): packed weights, biases, h ping-pong x2,
  // barrier, h1/h2 sequence buffers (bf16).
  char* ws = (char*)d_ws;
  size_t off = 0;
  auto carve = [&](size_t bytes) -> void* {
    off = (off + 255) & ~(size_t)255;
    void* p = ws + off;
    off += bytes;
    return p;
  };
  unsigned short* W1    = (unsigned short*)carve((size_t)NROWS * KPAD1 * 2);
  unsigned short* W2    = (unsigned short*)carve((size_t)NROWS * KPAD2 * 2);
  float*          Bs1   = (float*)carve(NROWS * 4);
  float*          Bs2   = (float*)carve(NROWS * 4);
  unsigned short* hbuf1 = (unsigned short*)carve((size_t)2 * B_SZ * H_SZ * 2);
  unsigned short* hbuf2 = (unsigned short*)carve((size_t)2 * B_SZ * H_SZ * 2);
  unsigned int*   bar   = (unsigned int*)carve(256);
  unsigned short* h1    = (unsigned short*)carve((size_t)B_SZ * T_SZ * H_SZ * 2);
  unsigned short* h2    = (unsigned short*)carve((size_t)B_SZ * T_SZ * H_SZ * 2);

  init_kernel<<<1, 64, 0, stream>>>(bar);
  prepack_kernel<<<512, 256, 0, stream>>>(Wih1, Whh1, bih1, bhh1,
                                          Wih2, Whh2, bih2, bhh2,
                                          W1, W2, Bs1, Bs2);

  // LDS: max of the two layer footprints (layer2: 64KB A + 16KB gates + ...)
  const size_t smem = (size_t)B_SZ * KPAD2 * 2 +
                      (4 * B_SZ * JS + B_SZ * JS + 4 * JS) * 4;

  lstm2_kernel<<<NWG_TOT, NTHREADS, smem, stream>>>(
      x, W1, Bs1, W2, Bs2, h1, h2, hbuf1, hbuf2, bar);

  fc_kernel<<<(B_SZ * T_SZ + 255) / 256, 256, 0, stream>>>(h2, Wfc, bfc, out);
}
